// Decoder_81252191305810
// MI455X (gfx1250) — matrix operand order repaired and checked by reading, not compiled, not run on hardware
//
#include <hip/hip_runtime.h>
#include <hip/hip_bf16.h>
#include <cstddef>

// ---------------- problem constants ----------------
#define BQ      64
#define TENC    1024
#define INF     512
#define MELD    80
#define MEMIN   400      // MEL_D * R
#define MEMINP  416      // padded to multiple of 32
#define ATTND   128
#define RNND    256
#define NFILT   32
#define KCONV   31
#define NSTEPS  200
#define TCHUNK  128

typedef __attribute__((ext_vector_type(16))) __bf16 v16bf;
typedef __attribute__((ext_vector_type(8)))  __bf16 v8bf;   // 16 bytes
typedef __attribute__((ext_vector_type(8)))  float  v8f;
typedef __attribute__((ext_vector_type(4)))  float  f4;     // 16 bytes

__device__ __forceinline__ __bf16 f2bf(float x) {
    union { float f; unsigned u; } v; v.f = x;
    unsigned r = (v.u + 0x7fffu + ((v.u >> 16) & 1u)) >> 16;
    unsigned short h = (unsigned short)r;
    __bf16 o; __builtin_memcpy(&o, &h, 2);
    return o;
}

// pack two floats to packed bf16 pair (RNE) in one v_perm_b32
__device__ __forceinline__ unsigned pack2bf(float a, float b) {
    unsigned ua = __float_as_uint(a), ub = __float_as_uint(b);
    ua = ua + 0x7fffu + ((ua >> 16) & 1u);
    ub = ub + 0x7fffu + ((ub >> 16) & 1u);
    return __builtin_amdgcn_perm(ub, ua, 0x07060302u);  // {ub.hi16, ua.hi16}
}

__device__ __forceinline__ float frcp(float x) { return __builtin_amdgcn_rcpf(x); }
__device__ __forceinline__ float sigf(float x) { return frcp(1.f + __expf(-x)); }
// tanh(x) = 1 - 2/(e^{2x}+1); exp overflow -> +1, underflow -> -1 (correct limits)
__device__ __forceinline__ float tanh_fast(float x) {
    return 1.f - 2.f * frcp(__expf(2.f * x) + 1.f);
}

// ---- WMMA fragment helpers (16x16x32 bf16, wave32) ----
__device__ __forceinline__ v16bf frag_a_pack(const float* __restrict__ p) {
    // p already points at this lane's row + k-chunk + kb offset
    f4 x0 = *(const f4*)(p);        // k: kb..kb+3
    f4 x1 = *(const f4*)(p + 4);    // k: kb+4..kb+7
    f4 x2 = *(const f4*)(p + 16);   // k: 16+kb..
    f4 x3 = *(const f4*)(p + 20);
    union { v16bf v; unsigned u[8]; } rr;
    rr.u[0] = pack2bf(x0[0], x0[1]);
    rr.u[1] = pack2bf(x0[2], x0[3]);
    rr.u[2] = pack2bf(x1[0], x1[1]);
    rr.u[3] = pack2bf(x1[2], x1[3]);
    rr.u[4] = pack2bf(x2[0], x2[1]);
    rr.u[5] = pack2bf(x2[2], x2[3]);
    rr.u[6] = pack2bf(x3[0], x3[1]);
    rr.u[7] = pack2bf(x3[2], x3[3]);
    return rr.v;
}

__device__ __forceinline__ v16bf frag_b_w(const __bf16* __restrict__ W, int ldw,
                                          int n0, int k0, int lane) {
    int c  = n0 + (lane & 15);
    int kb = (lane & 16) ? 8 : 0;
    const __bf16* p = W + (size_t)c * ldw + k0 + kb;
    v8bf b0 = *(const v8bf*)(p);        // 16B
    v8bf b1 = *(const v8bf*)(p + 16);   // 16B
    return __builtin_shufflevector(b0, b1, 0, 1, 2, 3, 4, 5, 6, 7,
                                   8, 9, 10, 11, 12, 13, 14, 15);
}

// K-segment loop with fixed A base: strength-reduced pointer walk, 2 WMMA per chunk.
__device__ __forceinline__ void kloop_seg(const float* __restrict__ A, long lda,
                                          int m0, int n0,
                                          const __bf16* __restrict__ W, int ldw,
                                          int kw0, int kcnt, int lane,
                                          v8f& acc0, v8f& acc1) {
    const float* ap = A + (size_t)(m0 + (lane & 15)) * (size_t)lda + ((lane & 16) ? 8 : 0);
#pragma unroll 2
    for (int k = 0; k < kcnt; k += 32) {
        v16bf a  = frag_a_pack(ap + k);
        v16bf b0 = frag_b_w(W, ldw, n0,      kw0 + k, lane);
        v16bf b1 = frag_b_w(W, ldw, n0 + 16, kw0 + k, lane);
        acc0 = __builtin_amdgcn_wmma_f32_16x16x32_bf16(false, a, false, b0,
                                                       (short)0, acc0, false, false);
        acc1 = __builtin_amdgcn_wmma_f32_16x16x32_bf16(false, a, false, b1,
                                                       (short)0, acc1, false, false);
    }
}

// ---------------- generic GEMM: C = act([A|A2] @ W^T + bias) ----------------
// A supplies k in [0,Ksplit), A2 supplies k in [Ksplit,Kd); two sequential loops
// with fixed base pointers. W: Npad x Kd bf16 zero padded. Each wave computes a
// 16x32 tile group (A fragment reused for 2 WMMAs). Ksplit==Kd => single loop.
__global__ void gemm_bf16_kernel(const float* __restrict__ A, long lda,
                                 const float* __restrict__ A2, long lda2, int Ksplit,
                                 const __bf16* __restrict__ W, int ldw,
                                 const float* __restrict__ bias,
                                 float* __restrict__ C, long ldc,
                                 int M, int N, int Npad, int Kd, int act) {
    int wave = (int)((blockIdx.x * blockDim.x + threadIdx.x) >> 5);  // wave-uniform
    int lane = threadIdx.x & 31;
    int tn2 = Npad >> 5;
    int tiles = (M >> 4) * tn2;
    if (wave >= tiles) return;               // whole-wave exit: EXEC all-ones at WMMA
    int mt = wave / tn2, ng = wave - mt * tn2;
    int m0 = mt << 4, n0 = ng << 5;
    v8f acc0 = {}, acc1 = {};
    kloop_seg(A, lda, m0, n0, W, ldw, 0, Ksplit, lane, acc0, acc1);
    if (Kd > Ksplit)
        kloop_seg(A2, lda2, m0, n0, W, ldw, Ksplit, Kd - Ksplit, lane, acc0, acc1);
    int rb = (lane & 16) ? 8 : 0;
#pragma unroll
    for (int half = 0; half < 2; ++half) {
        int c = n0 + half * 16 + (lane & 15);
        v8f acc = half ? acc1 : acc0;
        if (c < N) {
            float bv = bias ? bias[c] : 0.f;
#pragma unroll
            for (int i = 0; i < 8; ++i) {
                float v = acc[i] + bv;
                if (act == 1)      v = v > 0.f ? v : 0.f;
                else if (act == 2) v = sigf(v);
                C[(size_t)(m0 + rb + i) * (size_t)ldc + c] = v;
            }
        }
    }
}

// ---------------- weight f32 -> bf16 convert with zero padding ----------------
__global__ void cvt_pad_kernel(const float* __restrict__ src, __bf16* __restrict__ dst,
                               int N, int K, int Npad, int Kpad) {
    int i = blockIdx.x * blockDim.x + threadIdx.x;
    if (i >= Npad * Kpad) return;
    int r = i / Kpad, c = i - r * Kpad;
    dst[i] = (r < N && c < K) ? f2bf(src[r * K + c]) : f2bf(0.f);
}

// ---------------- stage A with zero pad (lda==0 => broadcast row) ----------------
__global__ void stage_pad_kernel(const float* __restrict__ src, long lda,
                                 float* __restrict__ dst, int rows, int K, int Kpad) {
    int i = blockIdx.x * blockDim.x + threadIdx.x;
    if (i >= rows * Kpad) return;
    int r = i / Kpad, c = i - r * Kpad;
    dst[i] = (c < K) ? src[(size_t)r * (size_t)lda + c] : 0.f;
}

// ---------------- state init ----------------
__global__ void init_state_kernel(const float* __restrict__ attn_rnn_init,
                                  const float* __restrict__ dec_rnn_init,
                                  float* attnh, float* h1, float* h2,
                                  float* ctx, float* attn, float* cum) {
    int i = blockIdx.x * blockDim.x + threadIdx.x;
    if (i < BQ * TENC) { attn[i] = 0.f; cum[i] = 0.f; }
    if (i < BQ * INF)  ctx[i] = 0.f;
    if (i < BQ * RNND) {
        int j = i & (RNND - 1);
        attnh[i] = attn_rnn_init[j];
        h1[i] = dec_rnn_init[j];
        h2[i] = dec_rnn_init[RNND + j];
    }
}

// ---------------- GRU gate combine (optionally d += h_new) ----------------
__global__ void gru_combine_kernel(const float* __restrict__ gi, const float* __restrict__ gh,
                                   float* __restrict__ h, float* __restrict__ dres, int addres) {
    int i = blockIdx.x * blockDim.x + threadIdx.x;
    if (i >= BQ * RNND) return;
    int b = i >> 8, j = i & 255;
    const float* gib = gi + b * (3 * RNND);
    const float* ghb = gh + b * (3 * RNND);
    float r = sigf(gib[j] + ghb[j]);
    float z = sigf(gib[RNND + j] + ghb[RNND + j]);
    float n = tanh_fast(gib[2 * RNND + j] + r * ghb[2 * RNND + j]);
    float hn = (1.f - z) * n + z * h[i];
    h[i] = hn;
    if (addres) dres[i] += hn;
}

// ---------------- fused location conv + pl WMMA + energy ----------------
// grid (TENC/TCHUNK, BQ), 256 threads. sig[b,t] = sigmoid(e) * mask
__global__ void attn_energy_kernel(const float* __restrict__ attn,
                                   const float* __restrict__ attn_cum,
                                   const float* __restrict__ convw,     // 32 x 2 x 31 f32
                                   const __bf16* __restrict__ wloclin,  // 128 x 32 bf16
                                   const float* __restrict__ bloclin,   // 128
                                   const float* __restrict__ q,         // B x 128
                                   const float* __restrict__ pa,        // B x T x 128
                                   const float* __restrict__ vw,        // 128
                                   const unsigned char* __restrict__ mask,
                                   float* __restrict__ sig) {
    int b  = blockIdx.y;
    int t0 = blockIdx.x * TCHUNK;
    __shared__ float sA[TCHUNK + KCONV - 1], sC[TCHUNK + KCONV - 1];
    __shared__ __align__(16) __bf16 sconv[TCHUNK][NFILT];   // 8 KB WMMA A staging
    __shared__ float sE[TCHUNK];
    __shared__ float sQ[ATTND], sLB[ATTND], sV[ATTND];
    int tid = threadIdx.x;

    for (int i = tid; i < TCHUNK + KCONV - 1; i += 256) {
        int t = t0 + i - (KCONV - 1) / 2;
        bool ok = (t >= 0 && t < TENC);
        sA[i] = ok ? attn[b * TENC + t] : 0.f;
        sC[i] = ok ? attn_cum[b * TENC + t] : 0.f;
    }
    if (tid < ATTND) {
        sQ[tid] = q[b * ATTND + tid];
        sLB[tid] = bloclin[tid];
        sV[tid] = vw[tid];
        sE[tid] = 0.f;
    }
    __syncthreads();

    // location conv: lconv[t,f]
    for (int o = tid; o < TCHUNK * NFILT; o += 256) {
        int f = o & (NFILT - 1), tl = o >> 5;
        const float* wa = convw + f * (2 * KCONV);
        const float* wc = wa + KCONV;
        float acc = 0.f;
#pragma unroll
        for (int k = 0; k < KCONV; ++k) acc += wa[k] * sA[tl + k] + wc[k] * sC[tl + k];
        sconv[tl][f] = f2bf(acc);
    }
    __syncthreads();

    // pl = lconv @ wloclin^T via WMMA (K=32, single op), fused energy reduction
    int wave = tid >> 5, lane = tid & 31;
    for (int tile = wave; tile < (TCHUNK / 16) * (ATTND / 16); tile += 8) {
        int mt = tile >> 3, nt = tile & 7;
        int m0 = mt << 4, n0 = nt << 4;
        int kb = (lane & 16) ? 8 : 0;
        int r  = m0 + (lane & 15);
        const __bf16* ap = &sconv[r][kb];
        v8bf a0 = *(const v8bf*)(ap);        // f: kb..kb+7
        v8bf a1 = *(const v8bf*)(ap + 16);   // f: 16+kb..
        v16bf a = __builtin_shufflevector(a0, a1, 0, 1, 2, 3, 4, 5, 6, 7,
                                          8, 9, 10, 11, 12, 13, 14, 15);
        v16bf wb = frag_b_w(wloclin, NFILT, n0, 0, lane);
        v8f acc = {};
        acc = __builtin_amdgcn_wmma_f32_16x16x32_bf16(false, a, false, wb,
                                                      (short)0, acc, false, false);
        int c  = n0 + (lane & 15);
        int rb = (lane & 16) ? 8 : 0;
        float qv = sQ[c], lb = sLB[c], vv = sV[c];
        const float* pab = pa + ((size_t)b * TENC + t0) * ATTND + c;
#pragma unroll
        for (int i = 0; i < 8; ++i) {
            int tl = m0 + rb + i;
            float x = qv + lb + acc[i] + pab[(size_t)tl * ATTND];
            atomicAdd(&sE[tl], vv * tanh_fast(x));       // ds_add_f32
        }
    }
    __syncthreads();
    if (tid < TCHUNK) {
        int t = t0 + tid;
        float m = mask[b * TENC + t] ? 1.f : 0.f;
        sig[b * TENC + t] = sigf(sE[tid]) * m;
    }
}

// ---------------- normalize + context GEMV + carry update ----------------
__global__ void attn_ctx_kernel(const float* __restrict__ sig,
                                const float* __restrict__ inputs,
                                float* __restrict__ attn, float* __restrict__ cum,
                                float* __restrict__ ctx, float* __restrict__ align_out) {
    int b = blockIdx.x, tid = threadIdx.x;  // 256 threads
    __shared__ float sAl[TENC];
    __shared__ float red[256];
    __shared__ f4 red4[128];
    float part = 0.f;
    for (int t = tid; t < TENC; t += 256) { float v = sig[b * TENC + t]; sAl[t] = v; part += v; }
    red[tid] = part;
    __syncthreads();
    for (int o = 128; o > 0; o >>= 1) { if (tid < o) red[tid] += red[tid + o]; __syncthreads(); }
    float inv = frcp(red[0]);
    for (int t = tid; t < TENC; t += 256) {
        float al = sAl[t] * inv;
        sAl[t] = al;
        attn[b * TENC + t] = al;
        cum[b * TENC + t] += al;
        align_out[(size_t)b * (NSTEPS * TENC) + t] = al;
    }
    __syncthreads();
    // ctx[b,d] = sum_t align[t]*inputs[b,t,d]: b128 loads, 2-way split-K over t.
    int dg = (tid & 127) << 2;           // d in {0,4,...,508}
    int th = tid >> 7;                   // t phase 0/1
    f4 acc = {0.f, 0.f, 0.f, 0.f};
    const float* ip = inputs + (size_t)b * TENC * INF + dg;
    for (int t = th; t < TENC; t += 2) {
        f4 x = *(const f4*)(ip + (size_t)t * INF);
        float al = sAl[t];
        acc[0] += al * x[0]; acc[1] += al * x[1];
        acc[2] += al * x[2]; acc[3] += al * x[3];
    }
    if (th == 1) red4[tid & 127] = acc;
    __syncthreads();
    if (th == 0) {
        f4 o = red4[tid];
        float* cp = ctx + b * INF + dg;
        cp[0] = acc[0] + o[0]; cp[1] = acc[1] + o[1];
        cp[2] = acc[2] + o[2]; cp[3] = acc[3] + o[3];
    }
}

// ---------------- stop token ----------------
__global__ void stop_kernel(const float* __restrict__ d,        // B x 256
                            const float* __restrict__ outs,     // d_out + s*400, stride 200*400
                            const float* __restrict__ stop_w, const float* __restrict__ stop_b,
                            float* __restrict__ stops) {        // d_out + STOP + s, stride 200
    int b = blockIdx.x, tid = threadIdx.x;
    __shared__ float red[256];
    float acc = 0.f;
    for (int j = tid; j < RNND + MEMIN; j += 256) {
        float x = (j < RNND) ? d[b * RNND + j]
                             : outs[(size_t)b * (NSTEPS * MEMIN) + (j - RNND)];
        acc += x * stop_w[j];
    }
    red[tid] = acc;
    __syncthreads();
    for (int o = 128; o > 0; o >>= 1) { if (tid < o) red[tid] += red[tid + o]; __syncthreads(); }
    if (tid == 0) stops[(size_t)b * NSTEPS] = sigf(red[0] + stop_b[0]);
}

// =======================================================================
extern "C" void kernel_launch(void* const* d_in, const int* in_sizes, int n_in,
                              void* d_out, int out_size, void* d_ws, size_t ws_size,
                              hipStream_t stream) {
    (void)in_sizes; (void)n_in; (void)out_size; (void)ws_size;
    const float* inputs = (const float*)d_in[0];
    const float* memory = (const float*)d_in[1];
    const unsigned char* mask = (const unsigned char*)d_in[2];
    int i = 3;
    const float* prenet_w1 = (const float*)d_in[i++];
    const float* prenet_b1 = (const float*)d_in[i++];
    const float* prenet_w2 = (const float*)d_in[i++];
    const float* prenet_b2 = (const float*)d_in[i++];
    const float* agru_wih  = (const float*)d_in[i++];
    const float* agru_whh  = (const float*)d_in[i++];
    const float* agru_bih  = (const float*)d_in[i++];
    const float* agru_bhh  = (const float*)d_in[i++];
    const float* loc_conv_w= (const float*)d_in[i++];
    const float* loc_lin_w = (const float*)d_in[i++];
    const float* loc_lin_b = (const float*)d_in[i++];
    const float* query_w   = (const float*)d_in[i++];
    const float* query_b   = (const float*)d_in[i++];
    const float* annot_w   = (const float*)d_in[i++];
    const float* annot_b   = (const float*)d_in[i++];
    const float* v_w       = (const float*)d_in[i++];
    const float* proj_w    = (const float*)d_in[i++];
    const float* proj_b    = (const float*)d_in[i++];
    const float* dec1_wih  = (const float*)d_in[i++];
    const float* dec1_whh  = (const float*)d_in[i++];
    const float* dec1_bih  = (const float*)d_in[i++];
    const float* dec1_bhh  = (const float*)d_in[i++];
    const float* dec2_wih  = (const float*)d_in[i++];
    const float* dec2_whh  = (const float*)d_in[i++];
    const float* dec2_bih  = (const float*)d_in[i++];
    const float* dec2_bhh  = (const float*)d_in[i++];
    const float* mel_w     = (const float*)d_in[i++];
    const float* mel_b     = (const float*)d_in[i++];
    const float* stop_w    = (const float*)d_in[i++];
    const float* stop_b    = (const float*)d_in[i++];
    const float* attn_rnn_init = (const float*)d_in[i++];
    const float* memory_init   = (const float*)d_in[i++];
    const float* dec_rnn_init  = (const float*)d_in[i++];

    float* out_f = (float*)d_out;
    const size_t OUT_ALIGN = (size_t)BQ * NSTEPS * MEMIN;               // 5,120,000
    const size_t OUT_STOP  = OUT_ALIGN + (size_t)BQ * NSTEPS * TENC;    // 18,227,200

    // ---- workspace: bf16 weights (zero padded) then f32 buffers ----
    __bf16* wbp = (__bf16*)d_ws;
    size_t o = 0;
    __bf16* w_pre1  = wbp + o; o += (size_t)256 * MEMINP;   // K padded 400->416
    __bf16* w_pre2  = wbp + o; o += (size_t)128 * 256;
    __bf16* w_awih  = wbp + o; o += (size_t)768 * 640;
    __bf16* w_awhh  = wbp + o; o += (size_t)768 * 256;
    __bf16* w_locl  = wbp + o; o += (size_t)ATTND * NFILT;
    __bf16* w_query = wbp + o; o += (size_t)ATTND * 256;
    __bf16* w_annot = wbp + o; o += (size_t)ATTND * INF;
    __bf16* w_proj  = wbp + o; o += (size_t)256 * 768;
    __bf16* w_d1wih = wbp + o; o += (size_t)768 * 256;
    __bf16* w_d1whh = wbp + o; o += (size_t)768 * 256;
    __bf16* w_d2wih = wbp + o; o += (size_t)768 * 256;
    __bf16* w_d2whh = wbp + o; o += (size_t)768 * 256;
    __bf16* w_mel   = wbp + o; o += (size_t)MEMINP * 256;   // N padded 400->416
    size_t bf_bytes = (o * 2 + 255) & ~(size_t)255;

    float* fb = (float*)((char*)d_ws + bf_bytes);
    size_t fo = 0;
    float* pa    = fb + fo; fo += (size_t)BQ * TENC * ATTND;
    float* xpad  = fb + fo; fo += (size_t)BQ * MEMINP;
    float* pm1   = fb + fo; fo += (size_t)BQ * 256;
    float* pm    = fb + fo; fo += (size_t)BQ * 128;
    float* gi    = fb + fo; fo += (size_t)BQ * 768;
    float* gh    = fb + fo; fo += (size_t)BQ * 768;
    float* attnh = fb + fo; fo += (size_t)BQ * RNND;
    float* qb    = fb + fo; fo += (size_t)BQ * ATTND;
    float* sg    = fb + fo; fo += (size_t)BQ * TENC;
    float* attn  = fb + fo; fo += (size_t)BQ * TENC;
    float* cum   = fb + fo; fo += (size_t)BQ * TENC;
    float* ctx   = fb + fo; fo += (size_t)BQ * INF;
    float* dbuf  = fb + fo; fo += (size_t)BQ * RNND;
    float* h1    = fb + fo; fo += (size_t)BQ * RNND;
    float* h2    = fb + fo; fo += (size_t)BQ * RNND;

    auto cvt = [&](const float* s, __bf16* d, int N, int K, int Npad, int Kpad) {
        int n = Npad * Kpad;
        cvt_pad_kernel<<<(n + 255) / 256, 256, 0, stream>>>(s, d, N, K, Npad, Kpad);
    };
    cvt(prenet_w1, w_pre1, 256, MEMIN, 256, MEMINP);
    cvt(prenet_w2, w_pre2, 128, 256, 128, 256);
    cvt(agru_wih,  w_awih, 768, 640, 768, 640);
    cvt(agru_whh,  w_awhh, 768, 256, 768, 256);
    cvt(loc_lin_w, w_locl, ATTND, NFILT, ATTND, NFILT);
    cvt(query_w,   w_query, ATTND, 256, ATTND, 256);
    cvt(annot_w,   w_annot, ATTND, INF, ATTND, INF);
    cvt(proj_w,    w_proj, 256, 768, 256, 768);
    cvt(dec1_wih,  w_d1wih, 768, 256, 768, 256);
    cvt(dec1_whh,  w_d1whh, 768, 256, 768, 256);
    cvt(dec2_wih,  w_d2wih, 768, 256, 768, 256);
    cvt(dec2_whh,  w_d2whh, 768, 256, 768, 256);
    cvt(mel_w,     w_mel, MEMIN, 256, MEMINP, 256);

    init_state_kernel<<<(BQ * TENC + 255) / 256, 256, 0, stream>>>(
        attn_rnn_init, dec_rnn_init, attnh, h1, h2, ctx, attn, cum);

    // single-source GEMM (Ksplit == Kd skips the second loop)
    auto gemm = [&](const float* A, long lda, const __bf16* W, int ldw, const float* bias,
                    float* C, long ldc, int M, int N, int Npad, int Kd, int act) {
        int tiles = (M / 16) * (Npad / 32);
        int blocks = (tiles + 7) / 8;
        gemm_bf16_kernel<<<blocks, 256, 0, stream>>>(A, lda, A, lda, Kd, W, ldw, bias,
                                                     C, ldc, M, N, Npad, Kd, act);
    };
    // split-A GEMM (fused concat [A | A2] along K)
    auto gemmS = [&](const float* A, long lda, const float* A2, long lda2, int Ksplit,
                     const __bf16* W, int ldw, const float* bias,
                     float* C, long ldc, int M, int N, int Npad, int Kd, int act) {
        int tiles = (M / 16) * (Npad / 32);
        int blocks = (tiles + 7) / 8;
        gemm_bf16_kernel<<<blocks, 256, 0, stream>>>(A, lda, A2, lda2, Ksplit, W, ldw, bias,
                                                     C, ldc, M, N, Npad, Kd, act);
    };

    // pa = inputs @ annot_w^T + annot_b   (one-time 8.6 GFLOP WMMA GEMM)
    gemm(inputs, INF, w_annot, INF, annot_b, pa, ATTND, BQ * TENC, ATTND, ATTND, INF, 0);

    for (int s = 0; s < NSTEPS; ++s) {
        // prenet (teacher forcing: frame s-1, or memory_init broadcast at s==0)
        const float* memin = (s == 0) ? memory_init : (memory + (size_t)(s - 1) * MEMIN);
        long lda0 = (s == 0) ? 0 : (long)(1000 * MELD);
        stage_pad_kernel<<<(BQ * MEMINP + 255) / 256, 256, 0, stream>>>(
            memin, lda0, xpad, BQ, MEMIN, MEMINP);
        gemm(xpad, MEMINP, w_pre1, MEMINP, prenet_b1, pm1, 256, BQ, 256, 256, MEMINP, 1);
        gemm(pm1, 256, w_pre2, 256, prenet_b2, pm, 128, BQ, 128, 128, 256, 1);

        // attention GRU: x = [pm | ctx] fused via split-A
        gemmS(pm, 128, ctx, INF, 128, w_awih, 640, agru_bih, gi, 768, BQ, 768, 768, 640, 0);
        gemm(attnh, 256, w_awhh, 256, agru_bhh, gh, 768, BQ, 768, 768, 256, 0);
        gru_combine_kernel<<<(BQ * RNND + 255) / 256, 256, 0, stream>>>(gi, gh, attnh, nullptr, 0);

        // query + location-sensitive attention
        gemm(attnh, 256, w_query, 256, query_b, qb, ATTND, BQ, ATTND, ATTND, 256, 0);
        attn_energy_kernel<<<dim3(TENC / TCHUNK, BQ), 256, 0, stream>>>(
            attn, cum, loc_conv_w, w_locl, loc_lin_b, qb, pa, v_w, mask, sg);
        attn_ctx_kernel<<<BQ, 256, 0, stream>>>(
            sg, inputs, attn, cum, ctx, out_f + OUT_ALIGN + (size_t)s * TENC);

        // decoder stack: d = [attnh | ctx] @ proj^T fused via split-A
        gemmS(attnh, 256, ctx, INF, 256, w_proj, 768, proj_b, dbuf, 256, BQ, 256, 256, 768, 0);
        gemm(dbuf, 256, w_d1wih, 256, dec1_bih, gi, 768, BQ, 768, 768, 256, 0);
        gemm(h1, 256, w_d1whh, 256, dec1_bhh, gh, 768, BQ, 768, 768, 256, 0);
        gru_combine_kernel<<<(BQ * RNND + 255) / 256, 256, 0, stream>>>(gi, gh, h1, dbuf, 1);
        gemm(dbuf, 256, w_d2wih, 256, dec2_bih, gi, 768, BQ, 768, 768, 256, 0);
        gemm(h2, 256, w_d2whh, 256, dec2_bhh, gh, 768, BQ, 768, 768, 256, 0);
        gru_combine_kernel<<<(BQ * RNND + 255) / 256, 256, 0, stream>>>(gi, gh, h2, dbuf, 1);

        // mel head (N padded 416, store guarded to 400; writes into outs slice)
        gemm(dbuf, 256, w_mel, 256, mel_b, out_f + (size_t)s * MEMIN,
             (long)NSTEPS * MEMIN, BQ, MEMIN, MEMINP, 256, 2);
        stop_kernel<<<BQ, 256, 0, stream>>>(dbuf, out_f + (size_t)s * MEMIN,
                                            stop_w, stop_b, out_f + OUT_STOP + s);
    }
}
